// ObverterReceiver_60060822667893
// MI455X (gfx1250) — matrix operand order repaired and checked by reading, not compiled, not run on hardware
//
#include <hip/hip_runtime.h>
#include <hip/hip_bf16.h>
#include <math.h>

#define B_    1024
#define S_    64
#define E_    256
#define H_    512
#define G4_   2048   // 4*H
#define KC_   768    // E + H
#define META_ 13

typedef __attribute__((ext_vector_type(16))) __bf16 v16bf;
typedef __attribute__((ext_vector_type(8)))  __bf16 v8bf;
typedef __attribute__((ext_vector_type(8)))  float  v8f;

static __device__ __forceinline__ float sigf(float x) { return 1.0f / (1.0f + expf(-x)); }

// ---------------- embedding gather: emb[b,s,:] = table[msg[b,s],:] (f32 -> d_out) ----
__global__ __launch_bounds__(256) void k_gather(const int* __restrict__ msg,
                                                const float* __restrict__ tab,
                                                float* __restrict__ out_emb) {
  size_t idx = (size_t)blockIdx.x * 256 + threadIdx.x;   // B*S*E
  int e   = (int)(idx & (E_ - 1));
  int row = (int)(idx >> 8);                              // b*S + s
  int v   = msg[row];
  out_emb[idx] = tab[(size_t)v * E_ + e];
}

// ---------------- pack bf16 combined weights Wc[j][0:256]=w_ih, Wc[j][256:768]=w_hh ---
__global__ __launch_bounds__(256) void k_conv_w(const float* __restrict__ w_ih,
                                                const float* __restrict__ w_hh,
                                                __bf16* __restrict__ Wc) {
  int idx = blockIdx.x * 256 + threadIdx.x;               // 2048*768
  int j = idx / KC_;
  int k = idx - j * KC_;
  float v = (k < E_) ? w_ih[j * E_ + k] : w_hh[j * H_ + (k - E_)];
  Wc[idx] = (__bf16)v;
}

__global__ __launch_bounds__(256) void k_conv_w1(const float* __restrict__ w1,
                                                 __bf16* __restrict__ Wc1) {
  int idx = blockIdx.x * 256 + threadIdx.x;               // 512*1024
  Wc1[idx] = (__bf16)w1[idx];
}

__global__ __launch_bounds__(256) void k_init(float* __restrict__ c, float* __restrict__ h,
                                              __bf16* __restrict__ hbf0,
                                              __bf16* __restrict__ hbf1) {
  int idx = blockIdx.x * 256 + threadIdx.x;               // B*H
  c[idx] = 0.0f; h[idx] = 0.0f;
  hbf0[idx] = (__bf16)0.0f; hbf1[idx] = (__bf16)0.0f;
}

// ---------------- fused LSTM step: gates GEMM (bf16 WMMA) + in-register cell update --
// wave = (mtile 0..63, n-slice 0..31): 4 accumulators at gate cols {n, H+n, 2H+n, 3H+n}
// so i,f,g,o for a (row, unit) pair live in the same lane/VGPR across accumulators.
// All four gate weight blocks are addressed from ONE base pointer with compile-time
// immediate offsets (g*H*KC elements, max ~2.4MB bytes < 24-bit IOFFSET) so the
// compiler keeps a single address register and clause-batches the B loads.
// 2048 waves = 256 blocks * 8 waves.
__global__ __launch_bounds__(256) void k_step(const float* __restrict__ emb, int s,
                                              const __bf16* __restrict__ hin,
                                              const __bf16* __restrict__ Wc,
                                              const float* __restrict__ b_ih,
                                              const float* __restrict__ b_hh,
                                              float* __restrict__ c,
                                              float* __restrict__ h,
                                              __bf16* __restrict__ hout) {
  int wave = blockIdx.x * 8 + (threadIdx.x >> 5);
  int lane = threadIdx.x & 31;
  int l15  = lane & 15;
  int half = lane >> 4;
  int mtile = wave & 63;        // 64 M tiles (batch)
  int nsl   = wave >> 6;        // 32 hidden slices of 16
  int m0 = mtile * 16, n0 = nsl * 16;
  int mrow = m0 + l15;
  int c0 = half * 16;           // A K-subchunk select
  int kb = half * 16;           // B K-half select

  const float*  arow  = emb + ((size_t)mrow * S_ + s) * E_;
  const __bf16* hrow  = hin + (size_t)mrow * H_;
  const __bf16* wbase = Wc  + (size_t)(n0 + l15) * KC_ + kb;   // gate 0 row; others at +g*H*KC

  v8f ai = {0,0,0,0,0,0,0,0}, af = {0,0,0,0,0,0,0,0};
  v8f ag = {0,0,0,0,0,0,0,0}, ao = {0,0,0,0,0,0,0,0};

  // phase 1: x_t (f32 -> bf16 in-register) against w_ih columns
  #pragma unroll
  for (int k0 = 0; k0 < E_; k0 += 32) {
    v16bf a;
    const float* p0 = arow + k0 + c0;
    const float* p1 = arow + k0 + 8 + c0;
    #pragma unroll
    for (int e = 0; e < 8; ++e) { a[e] = (__bf16)p0[e]; a[8 + e] = (__bf16)p1[e]; }
    v16bf bi_ = *reinterpret_cast<const v16bf*>(wbase + 0 * (H_ * KC_) + k0);
    v16bf bf_ = *reinterpret_cast<const v16bf*>(wbase + 1 * (H_ * KC_) + k0);
    v16bf bg_ = *reinterpret_cast<const v16bf*>(wbase + 2 * (H_ * KC_) + k0);
    v16bf bo_ = *reinterpret_cast<const v16bf*>(wbase + 3 * (H_ * KC_) + k0);
    ai = __builtin_amdgcn_wmma_f32_16x16x32_bf16(false, a, false, bi_, (short)0, ai, false, false);
    af = __builtin_amdgcn_wmma_f32_16x16x32_bf16(false, a, false, bf_, (short)0, af, false, false);
    ag = __builtin_amdgcn_wmma_f32_16x16x32_bf16(false, a, false, bg_, (short)0, ag, false, false);
    ao = __builtin_amdgcn_wmma_f32_16x16x32_bf16(false, a, false, bo_, (short)0, ao, false, false);
  }
  // phase 2: h_{t-1} (bf16) against w_hh columns
  #pragma unroll
  for (int k0 = 0; k0 < H_; k0 += 32) {
    v8bf lo = *reinterpret_cast<const v8bf*>(hrow + k0 + c0);
    v8bf hi = *reinterpret_cast<const v8bf*>(hrow + k0 + 8 + c0);
    v16bf a = __builtin_shufflevector(lo, hi, 0,1,2,3,4,5,6,7,8,9,10,11,12,13,14,15);
    v16bf bi_ = *reinterpret_cast<const v16bf*>(wbase + 0 * (H_ * KC_) + E_ + k0);
    v16bf bf_ = *reinterpret_cast<const v16bf*>(wbase + 1 * (H_ * KC_) + E_ + k0);
    v16bf bg_ = *reinterpret_cast<const v16bf*>(wbase + 2 * (H_ * KC_) + E_ + k0);
    v16bf bo_ = *reinterpret_cast<const v16bf*>(wbase + 3 * (H_ * KC_) + E_ + k0);
    ai = __builtin_amdgcn_wmma_f32_16x16x32_bf16(false, a, false, bi_, (short)0, ai, false, false);
    af = __builtin_amdgcn_wmma_f32_16x16x32_bf16(false, a, false, bf_, (short)0, af, false, false);
    ag = __builtin_amdgcn_wmma_f32_16x16x32_bf16(false, a, false, bg_, (short)0, ag, false, false);
    ao = __builtin_amdgcn_wmma_f32_16x16x32_bf16(false, a, false, bo_, (short)0, ao, false, false);
  }

  // in-register LSTM cell (PyTorch gate order i,f,g,o); lane owns hidden unit n.
  int n = n0 + l15;
  float bi = b_ih[n]          + b_hh[n];
  float bf = b_ih[H_ + n]     + b_hh[H_ + n];
  float bg = b_ih[2 * H_ + n] + b_hh[2 * H_ + n];
  float bo = b_ih[3 * H_ + n] + b_hh[3 * H_ + n];

  #pragma unroll
  for (int r = 0; r < 8; ++r) {
    int row = m0 + r + half * 8;
    size_t idx = (size_t)row * H_ + n;
    float i_ = sigf(ai[r] + bi);
    float f_ = sigf(af[r] + bf);
    float g_ = tanhf(ag[r] + bg);
    float o_ = sigf(ao[r] + bo);
    float cn = f_ * c[idx] + i_ * g_;
    c[idx] = cn;
    float hn = o_ * tanhf(cn);
    h[idx] = hn;
    hout[idx] = (__bf16)hn;
  }
}

// ---------------- img = image_representation @ proc_w^T + proc_b (K=13, tiny) --------
__global__ __launch_bounds__(256) void k_img(const float* __restrict__ ir,
                                             const float* __restrict__ pw,
                                             const float* __restrict__ pb,
                                             float* __restrict__ img) {
  int idx = blockIdx.x * 256 + threadIdx.x;               // B*H
  int b = idx >> 9;
  int n = idx & (H_ - 1);
  const float* x = ir + b * META_;
  const float* w = pw + n * META_;
  float acc = pb[n];
  #pragma unroll
  for (int m = 0; m < META_; ++m) acc += x[m] * w[m];
  img[idx] = acc;
}

// ---------------- hidden = relu([h | img] @ w1^T + b1) via bf16 WMMA -----------------
// 512 waves = 64 blocks * 8 waves; wave tile M=16, N=64; K=1024.
__global__ __launch_bounds__(256) void k_hidden(const float* __restrict__ h,
                                                const float* __restrict__ img,
                                                const __bf16* __restrict__ Wc1,
                                                const float* __restrict__ b1,
                                                float* __restrict__ hidden) {
  int wave = blockIdx.x * 8 + (threadIdx.x >> 5);
  int lane = threadIdx.x & 31;
  int l15  = lane & 15;
  int half = lane >> 4;
  int mtile = wave & 63;
  int ntile = wave >> 6;                                  // 0..7
  int m0 = mtile * 16, n0 = ntile * 64;
  int mrow = m0 + l15;
  int c0 = half * 16, kb = half * 16;

  const float*  hrow = h   + (size_t)mrow * H_;
  const float*  irow = img + (size_t)mrow * H_;
  const __bf16* w0   = Wc1 + (size_t)(n0 + l15) * (2 * H_) + kb;

  v8f acc0 = {0,0,0,0,0,0,0,0}, acc1 = {0,0,0,0,0,0,0,0};
  v8f acc2 = {0,0,0,0,0,0,0,0}, acc3 = {0,0,0,0,0,0,0,0};

  #pragma unroll
  for (int k0 = 0; k0 < 2 * H_; k0 += 32) {
    const float* src = (k0 < H_) ? (hrow + k0) : (irow + (k0 - H_));
    v16bf a;
    const float* p0 = src + c0;
    const float* p1 = src + 8 + c0;
    #pragma unroll
    for (int e = 0; e < 8; ++e) { a[e] = (__bf16)p0[e]; a[8 + e] = (__bf16)p1[e]; }
    v16bf b0 = *reinterpret_cast<const v16bf*>(w0 + 0 * 16 * (2 * H_) + k0);
    v16bf b1 = *reinterpret_cast<const v16bf*>(w0 + 1 * 16 * (2 * H_) + k0);
    v16bf b2 = *reinterpret_cast<const v16bf*>(w0 + 2 * 16 * (2 * H_) + k0);
    v16bf b3 = *reinterpret_cast<const v16bf*>(w0 + 3 * 16 * (2 * H_) + k0);
    acc0 = __builtin_amdgcn_wmma_f32_16x16x32_bf16(false, a, false, b0, (short)0, acc0, false, false);
    acc1 = __builtin_amdgcn_wmma_f32_16x16x32_bf16(false, a, false, b1, (short)0, acc1, false, false);
    acc2 = __builtin_amdgcn_wmma_f32_16x16x32_bf16(false, a, false, b2, (short)0, acc2, false, false);
    acc3 = __builtin_amdgcn_wmma_f32_16x16x32_bf16(false, a, false, b3, (short)0, acc3, false, false);
  }

  #pragma unroll
  for (int r = 0; r < 8; ++r) {
    size_t row = (size_t)(m0 + r + half * 8);
    float* out = hidden + row * H_ + n0 + l15;
    out[0]  = fmaxf(acc0[r] + b1[n0 + l15],      0.0f);
    out[16] = fmaxf(acc1[r] + b1[n0 + 16 + l15], 0.0f);
    out[32] = fmaxf(acc2[r] + b1[n0 + 32 + l15], 0.0f);
    out[48] = fmaxf(acc3[r] + b1[n0 + 48 + l15], 0.0f);
  }
}

// ---------------- prediction = hidden @ w2^T + b2 (N=2) ------------------------------
__global__ __launch_bounds__(256) void k_pred(const float* __restrict__ hidden,
                                              const float* __restrict__ w2,
                                              const float* __restrict__ b2,
                                              float* __restrict__ pred) {
  int idx = blockIdx.x * 256 + threadIdx.x;               // B*2
  int b = idx >> 1;
  int j = idx & 1;
  const float* hr = hidden + (size_t)b * H_;
  const float* wr = w2 + j * H_;
  float acc = b2[j];
  #pragma unroll 8
  for (int k = 0; k < H_; ++k) acc += hr[k] * wr[k];
  pred[idx] = acc;
}

extern "C" void kernel_launch(void* const* d_in, const int* in_sizes, int n_in,
                              void* d_out, int out_size, void* d_ws, size_t ws_size,
                              hipStream_t stream) {
  const float* image_rep = (const float*)d_in[0];
  const int*   messages  = (const int*)d_in[1];
  const float* embedding = (const float*)d_in[2];
  const float* w_ih      = (const float*)d_in[3];
  const float* w_hh      = (const float*)d_in[4];
  const float* b_ih      = (const float*)d_in[5];
  const float* b_hh      = (const float*)d_in[6];
  const float* proc_w    = (const float*)d_in[7];
  const float* proc_b    = (const float*)d_in[8];
  const float* w1        = (const float*)d_in[9];
  const float* b1        = (const float*)d_in[10];
  const float* w2        = (const float*)d_in[11];
  const float* b2        = (const float*)d_in[12];

  float* pred = (float*)d_out;
  float* emb  = (float*)d_out + (size_t)B_ * 2;           // emb output region, also GEMM A input

  // workspace carve-out (aligned to 256B)
  char*  ws  = (char*)d_ws;
  size_t off = 0;
  auto carve = [&](size_t bytes) -> void* {
    void* p = ws + off;
    off = (off + bytes + 255) & ~(size_t)255;
    return p;
  };
  __bf16* Wc     = (__bf16*)carve((size_t)G4_ * KC_ * 2);        // packed [w_ih|w_hh] bf16
  __bf16* Wc1    = (__bf16*)carve((size_t)H_ * 2 * H_ * 2);      // w1 bf16
  float*  cst    = (float*) carve((size_t)B_ * H_ * 4);
  float*  hst    = (float*) carve((size_t)B_ * H_ * 4);
  __bf16* hbf0   = (__bf16*)carve((size_t)B_ * H_ * 2);          // h bf16 ping
  __bf16* hbf1   = (__bf16*)carve((size_t)B_ * H_ * 2);          // h bf16 pong
  float*  img    = (float*) carve((size_t)B_ * H_ * 4);
  float*  hidden = (float*) carve((size_t)B_ * H_ * 4);

  // 1) embedding gather (writes emb output; also the x-feed for the LSTM GEMM)
  k_gather<<<(B_ * S_ * E_) / 256, 256, 0, stream>>>(messages, embedding, emb);
  // 2) one-time per-launch weight packing to bf16
  k_conv_w<<<(G4_ * KC_) / 256, 256, 0, stream>>>(w_ih, w_hh, Wc);
  k_conv_w1<<<(H_ * 2 * H_) / 256, 256, 0, stream>>>(w1, Wc1);
  k_init<<<(B_ * H_) / 256, 256, 0, stream>>>(cst, hst, hbf0, hbf1);

  // 3) 64 fused recurrent steps: [x_t|h] WMMA GEMM + in-register LSTM cell.
  //    h(bf16) is ping-ponged so no wave reads a buffer another wave is writing.
  for (int s = 0; s < S_; ++s) {
    const __bf16* hin  = (s & 1) ? hbf1 : hbf0;
    __bf16*       hout = (s & 1) ? hbf0 : hbf1;
    k_step<<<256, 256, 0, stream>>>(emb, s, hin, Wc, b_ih, b_hh, cst, hst, hout);
  }

  // 4) head
  k_img<<<(B_ * H_) / 256, 256, 0, stream>>>(image_rep, proc_w, proc_b, img);
  k_hidden<<<64, 256, 0, stream>>>(hst, img, Wc1, b1, hidden);
  k_pred<<<(B_ * 2) / 256, 256, 0, stream>>>(hidden, w2, b2, pred);
}
